// MultiheadAttention_11390253269230
// MI455X (gfx1250) — matrix operand order repaired and checked by reading, not compiled, not run on hardware
//
#include <hip/hip_runtime.h>
#include <hip/hip_fp16.h>
#include <cstdint>

// ---------------------------------------------------------------------------
// MHA forward for MI455X (gfx1250, wave32, WMMA).
//   B=2, S=2048, E=768, H=12, D=64.  Outputs: o [B,S,E] then a_avg [B,S,S].
// f16 storage + V_WMMA_F32_16X16X32_F16 accumulation; scores live in LDS
// (194KB/WG, legal on CDNA5's 320KB LDS); v stored transposed [B,H,D,S] so
// every WMMA B-fragment is a contiguous b128 load; q tile staged to LDS with
// gfx1250 async-DMA.  GEMM core: plain per-k loads + __launch_bounds__(.,1)
// so the scheduler pipelines with counted s_wait_loadcnt instead of spilling.
// ---------------------------------------------------------------------------

typedef __attribute__((ext_vector_type(16))) _Float16 v16h;
typedef __attribute__((ext_vector_type(8)))  float    v8f;

#define BB 2
#define SS 2048
#define EE 768
#define HH 12
#define DD 64
#define SCALE 0.125f            // 1/sqrt(64)
#define M_TOT (BB * SS)         // 4096
#define N_QKV (3 * EE)          // 2304

// ---------------- WMMA fragment helpers (contiguous k = 16h + s order for A and B) --

// A-matrix 16x32 f16: lane h=lane>>4, m=lane&15.
// VGPR v holds K pair: k = h*16 + v*2.
__device__ __forceinline__ v16h load_A16(const _Float16* base, int ld, int lane) {
    const int h = (lane >> 4) & 1, m = lane & 15;
    const _Float16* rp = base + (size_t)m * ld;
    v16h a;
#pragma unroll
    for (int v = 0; v < 8; ++v) {
        const int k = h * 16 + v * 2;
        a[2 * v + 0] = rp[k + 0];
        a[2 * v + 1] = rp[k + 1];
    }
    return a;
}

// B-matrix 32x16 f16 where Bmat[k][n] = W[n][k] (W row-major [N,K], K contig).
// lanes 0-15 hold K=0..15 (V0:k0,1 .. V7:k14,15), lanes 16-31 hold K=16..31.
__device__ __forceinline__ v16h load_BT16(const _Float16* base, int ld, int lane) {
    const int h = (lane >> 4) & 1, n = lane & 15;
    const _Float16* rp = base + (size_t)n * ld + h * 16;
    v16h b;
#pragma unroll
    for (int v = 0; v < 8; ++v) {
        b[2 * v + 0] = rp[2 * v + 0];
        b[2 * v + 1] = rp[2 * v + 1];
    }
    return b;
}

__device__ __forceinline__ v8f wmma32(v16h a, v16h b, v8f c) {
    return __builtin_amdgcn_wmma_f32_16x16x32_f16(false, a, false, b,
                                                  (short)0, c, false, false);
}

// ---------------- Kernel 0: f32 -> f16 convert ------------------------------
__global__ void cvt_f16_kernel(const float* __restrict__ src,
                               _Float16* __restrict__ dst, int n) {
    int i = blockIdx.x * blockDim.x + threadIdx.x;
    if (i < n) dst[i] = (_Float16)src[i];
}

// ---------------- shared GEMM core: one wave -> 32x64 tile ------------------
// C[2][4]: 2 m-tiles x 4 n-tiles.  A row-major [M,K], W row-major [N,K].
// No manual pipelining: with launch_bounds headroom the scheduler overlaps
// the s_clause'd b128 loads against the WMMA pipe via counted waits.
__device__ __forceinline__ void gemm_wave_32x64(const _Float16* __restrict__ Ab,
                                                const _Float16* __restrict__ Wb,
                                                int K, int lane, v8f c[2][4]) {
    for (int k0 = 0; k0 < K; k0 += 32) {
        v16h a0 = load_A16(Ab + k0, K, lane);
        v16h a1 = load_A16(Ab + (size_t)16 * K + k0, K, lane);
        v16h b0 = load_BT16(Wb + (size_t)0  * K + k0, K, lane);
        v16h b1 = load_BT16(Wb + (size_t)16 * K + k0, K, lane);
        v16h b2 = load_BT16(Wb + (size_t)32 * K + k0, K, lane);
        v16h b3 = load_BT16(Wb + (size_t)48 * K + k0, K, lane);
        c[0][0] = wmma32(a0, b0, c[0][0]);
        c[0][1] = wmma32(a0, b1, c[0][1]);
        c[0][2] = wmma32(a0, b2, c[0][2]);
        c[0][3] = wmma32(a0, b3, c[0][3]);
        c[1][0] = wmma32(a1, b0, c[1][0]);
        c[1][1] = wmma32(a1, b1, c[1][1]);
        c[1][2] = wmma32(a1, b2, c[1][2]);
        c[1][3] = wmma32(a1, b3, c[1][3]);
    }
}

// ---------------- Kernel 1: QKV GEMM + bias + head scatter ------------------
// out(r,c) = xh[r,:].wh[c,:] + bias[c]; q pre-scaled; v stored TRANSPOSED.
// Block: 128 threads (4 waves); each wave computes 32 rows x 64 cols.
__global__ void __launch_bounds__(128, 1)
qkv_gemm_kernel(const _Float16* __restrict__ xh,   // [4096,768]
                const _Float16* __restrict__ wh,   // [2304,768]
                const float*    __restrict__ bias, // [2304]
                _Float16* __restrict__ qh,         // [B,H,S,D]
                _Float16* __restrict__ kh,         // [B,H,S,D]
                _Float16* __restrict__ vt) {       // [B,H,D,S]
    const int lane = threadIdx.x & 31;
    const int wave = threadIdx.x >> 5;
    const int m0 = blockIdx.y * 128 + wave * 32;
    const int n0 = blockIdx.x * 64;

    v8f c[2][4] = {};
    gemm_wave_32x64(xh + (size_t)m0 * EE, wh + (size_t)n0 * EE, EE, lane, c);

    const int h16 = (lane >> 4) & 1, nl = lane & 15;
#pragma unroll
    for (int mt = 0; mt < 2; ++mt) {
#pragma unroll
        for (int t = 0; t < 4; ++t) {
#pragma unroll
            for (int v = 0; v < 8; ++v) {
                const int r   = m0 + 16 * mt + v + 8 * h16; // [0,4096)
                const int col = n0 + 16 * t + nl;           // [0,2304)
                float val = c[mt][t][v] + bias[col];
                const int which = col / EE;                 // 0=q 1=k 2=v
                const int e     = col - which * EE;
                const int head  = e >> 6;
                const int d     = e & 63;
                const int bb    = r >> 11;
                const int s     = r & 2047;
                const size_t hb = ((size_t)bb * HH + head);
                if (which == 0)
                    qh[(hb * SS + s) * DD + d] = (_Float16)(val * SCALE);
                else if (which == 1)
                    kh[(hb * SS + s) * DD + d] = (_Float16)val;
                else
                    vt[(hb * DD + d) * SS + s] = (_Float16)val;  // transposed
            }
        }
    }
}

// ---------------- Kernel 2: attention per (b, head, 16-row q tile) ----------
// LDS: scores f32 [16][2048] (128KB) | probs f16 [16][2048] (64KB) | q f16 2KB
#define SMEM_S_BYTES (16 * 2048 * 4)
#define SMEM_P_BYTES (16 * 2048 * 2)
#define SMEM_Q_BYTES (16 * 64 * 2)
#define SMEM_TOTAL   (SMEM_S_BYTES + SMEM_P_BYTES + SMEM_Q_BYTES)

__global__ void __launch_bounds__(256, 1)
attn_kernel(const _Float16* __restrict__ qh,   // [B,H,S,D]
            const _Float16* __restrict__ kh,   // [B,H,S,D]
            const _Float16* __restrict__ vt,   // [B,H,D,S]
            const float*    __restrict__ mask, // [B,1,S,S]
            _Float16*       __restrict__ ctx,  // [B,S,E]
            float*          __restrict__ a_avg)// [B,S,S]
{
    extern __shared__ char smem[];
    float*    sS = (float*)smem;                                    // [16][2048]
    _Float16* sP = (_Float16*)(smem + SMEM_S_BYTES);                // [16][2048]
    _Float16* sQ = (_Float16*)(smem + SMEM_S_BYTES + SMEM_P_BYTES); // [16][64]

    const int tid  = threadIdx.x;          // 256 threads = 8 waves
    const int lane = tid & 31;
    const int wave = tid >> 5;
    const int q0   = blockIdx.x * 16;
    const int head = blockIdx.y;
    const int b    = blockIdx.z;

    const size_t hbase = ((size_t)b * HH + head) * SS * DD; // same for q/k/vt

    // ---- phase 0: async DMA q tile (2KB) into LDS (gfx1250 ASYNCcnt path) --
    if (tid < 128) {
        const unsigned lds_off =
            (unsigned)(uintptr_t)sQ + (unsigned)tid * 16u; // LDS byte offset
        const unsigned long long gaddr =
            (unsigned long long)(uintptr_t)(qh + hbase + (size_t)q0 * DD +
                                            (size_t)tid * 8);
        asm volatile("global_load_async_to_lds_b128 %0, %1, off"
                     :: "v"(lds_off), "v"(gaddr) : "memory");
    }
    asm volatile("s_wait_asynccnt 0x0" ::: "memory");
    __syncthreads();

    // ---- phase 1: scores = q.k^T + mask, into LDS (pipelined over j) ------
    {
        const v16h a0 = load_A16(sQ + 0,  DD, lane);
        const v16h a1 = load_A16(sQ + 32, DD, lane);
        const int h16 = (lane >> 4) & 1, nl = lane & 15;
        const _Float16* kb = kh + hbase;

        auto emit = [&](int j0, const v8f& c) {
#pragma unroll
            for (int v = 0; v < 8; ++v) {
                const int row = v + 8 * h16;
                const int col = j0 + nl;
                const float mval =
                    mask[((size_t)b * SS + (q0 + row)) * SS + col];
                sS[row * 2048 + col] = c[v] + mval;
            }
        };

        int j0 = wave * 256;
        v16h b0 = load_BT16(kb + (size_t)j0 * DD + 0,  DD, lane);
        v16h b1 = load_BT16(kb + (size_t)j0 * DD + 32, DD, lane);
        for (int t = 0; t < 15; ++t) {
            const int jn = j0 + 16;
            v16h nb0 = load_BT16(kb + (size_t)jn * DD + 0,  DD, lane);
            v16h nb1 = load_BT16(kb + (size_t)jn * DD + 32, DD, lane);
            v8f c = {};
            c = wmma32(a0, b0, c);
            c = wmma32(a1, b1, c);
            emit(j0, c);
            b0 = nb0; b1 = nb1; j0 = jn;
        }
        v8f c = {};
        c = wmma32(a0, b0, c);
        c = wmma32(a1, b1, c);
        emit(j0, c);
    }
    __syncthreads();

    // ---- phase 2: exact softmax; wave w owns rows 2w, 2w+1 ----------------
    for (int rr = 0; rr < 2; ++rr) {
        const int row = 2 * wave + rr;
        float* rowp = sS + row * 2048;
        float mx = -1e30f;
        for (int c2 = lane; c2 < 2048; c2 += 32) mx = fmaxf(mx, rowp[c2]);
#pragma unroll
        for (int off = 16; off > 0; off >>= 1)
            mx = fmaxf(mx, __shfl_xor(mx, off, 32));
        float sum = 0.f;
        for (int c2 = lane; c2 < 2048; c2 += 32) {
            float e = __expf(rowp[c2] - mx);
            rowp[c2] = e;
            sum += e;
        }
#pragma unroll
        for (int off = 16; off > 0; off >>= 1)
            sum += __shfl_xor(sum, off, 32);
        const float inv = 1.0f / sum;
        _Float16* prow = sP + row * 2048;
        for (int c2 = lane; c2 < 2048; c2 += 32)
            prow[c2] = (_Float16)(rowp[c2] * inv);
    }
    __syncthreads();

    // ---- phase 3: waves 0-3: o_tile = p @ v (vT contiguous-K B frags);
    //               waves 4-7: a_avg atomics ------------------------------
    if (wave < 4) {
        const int n0 = wave * 16;
        const _Float16* vtb = vt + hbase + (size_t)n0 * SS; // vT[n0][0]
        v8f c = {};
        v16h a  = load_A16(sP, 2048, lane);
        v16h bb = load_BT16(vtb, SS, lane);
        for (int k0 = 0; k0 < SS - 32; k0 += 32) {
            v16h na = load_A16(sP + k0 + 32, 2048, lane);
            v16h nb = load_BT16(vtb + k0 + 32, SS, lane);
            c = wmma32(a, bb, c);
            a = na; bb = nb;
        }
        c = wmma32(a, bb, c);
        const int h16 = (lane >> 4) & 1, nl = lane & 15;
#pragma unroll
        for (int v = 0; v < 8; ++v) {
            const int row = v + 8 * h16;
            const int col = n0 + nl;
            ctx[((size_t)b * SS + (q0 + row)) * EE + head * DD + col] =
                (_Float16)c[v];
        }
    } else {
        const int t = tid - 128;
        const float invH = 1.0f / (float)HH;
        for (int i = t; i < 16 * 2048; i += 128) {
            const int row = i >> 11, col = i & 2047;
            atomicAdd(&a_avg[((size_t)b * SS + (q0 + row)) * SS + col],
                      (float)sP[row * 2048 + col] * invH);
        }
    }
}

// ---------------- Kernel 3: output projection -------------------------------
__global__ void __launch_bounds__(128, 1)
proj_gemm_kernel(const _Float16* __restrict__ ctx, // [4096,768]
                 const _Float16* __restrict__ wh,  // [768,768]
                 const float*    __restrict__ bias,// [768]
                 float* __restrict__ out) {        // [4096,768]
    const int lane = threadIdx.x & 31;
    const int wave = threadIdx.x >> 5;
    const int m0 = blockIdx.y * 128 + wave * 32;
    const int n0 = blockIdx.x * 64;

    v8f c[2][4] = {};
    gemm_wave_32x64(ctx + (size_t)m0 * EE, wh + (size_t)n0 * EE, EE, lane, c);

    const int h16 = (lane >> 4) & 1, nl = lane & 15;
#pragma unroll
    for (int mt = 0; mt < 2; ++mt)
#pragma unroll
        for (int t = 0; t < 4; ++t)
#pragma unroll
            for (int v = 0; v < 8; ++v) {
                const int r   = m0 + 16 * mt + v + 8 * h16;
                const int col = n0 + 16 * t + nl;
                out[(size_t)r * EE + col] = c[mt][t][v] + bias[col];
            }
}

// ---------------- host launch ------------------------------------------------
extern "C" void kernel_launch(void* const* d_in, const int* in_sizes, int n_in,
                              void* d_out, int out_size, void* d_ws, size_t ws_size,
                              hipStream_t stream) {
    const float* x      = (const float*)d_in[0];  // [B,S,E]
    const float* mask   = (const float*)d_in[1];  // [B,1,S,S]
    const float* qkv_w  = (const float*)d_in[2];  // [3E,E]
    const float* qkv_b  = (const float*)d_in[3];  // [3E]
    const float* o_w    = (const float*)d_in[4];  // [E,E]
    const float* o_b    = (const float*)d_in[5];  // [E]

    float* out_o  = (float*)d_out;                         // [B,S,E]
    float* a_avg  = (float*)d_out + (size_t)BB * SS * EE;  // [B,S,S]

    // workspace carve-up (f16 staging buffers, ~27 MB total)
    char* ws = (char*)d_ws;
    _Float16* xh   = (_Float16*)(ws);                       // 4096*768
    _Float16* wqh  = (_Float16*)(ws + 6291456);             // 2304*768
    _Float16* owh  = (_Float16*)(ws + 6291456 + 3538944);   // 768*768
    _Float16* qh   = (_Float16*)(ws + 11010048);            // B*H*S*D
    _Float16* kh   = (_Float16*)(ws + 14155776);            // B*H*S*D
    _Float16* vt   = (_Float16*)(ws + 17301504);            // B*H*D*S
    _Float16* ctx  = (_Float16*)(ws + 20447232);            // 4096*768

    // f32 -> f16 conversions
    {
        int n;
        n = M_TOT * EE;
        cvt_f16_kernel<<<(n + 255) / 256, 256, 0, stream>>>(x, xh, n);
        n = N_QKV * EE;
        cvt_f16_kernel<<<(n + 255) / 256, 256, 0, stream>>>(qkv_w, wqh, n);
        n = EE * EE;
        cvt_f16_kernel<<<(n + 255) / 256, 256, 0, stream>>>(o_w, owh, n);
    }

    // zero a_avg (accumulated atomically) -- capture-legal stream memset
    hipMemsetAsync(a_avg, 0, (size_t)BB * SS * SS * sizeof(float), stream);

    // QKV projection: M=4096, N=2304, K=768 (wave tile 32x64)
    {
        dim3 grid(N_QKV / 64, M_TOT / 128);
        qkv_gemm_kernel<<<grid, 128, 0, stream>>>(xh, wqh, qkv_b, qh, kh, vt);
    }

    // attention: one block per (b, head, 16-row q tile); 194KB dynamic LDS
    {
        hipFuncSetAttribute((const void*)attn_kernel,
                            hipFuncAttributeMaxDynamicSharedMemorySize,
                            SMEM_TOTAL);
        dim3 grid(SS / 16, HH, BB);
        attn_kernel<<<grid, 256, SMEM_TOTAL, stream>>>(qh, kh, vt, mask, ctx,
                                                       a_avg);
    }

    // output projection: M=4096, N=768, K=768 (wave tile 32x64)
    {
        dim3 grid(EE / 64, M_TOT / 128);
        proj_gemm_kernel<<<grid, 128, 0, stream>>>(ctx, owh, o_b, out_o);
    }
}
